// HPPAcousticConvStack_80771154969149
// MI455X (gfx1250) — compile-verified, matrix operand order checked by reading, not run on hardware
//
#include <hip/hip_runtime.h>
#include <hip/hip_bf16.h>

typedef __attribute__((ext_vector_type(16))) _Float16 v16h;
typedef __attribute__((ext_vector_type(8)))  float    v8f;
typedef __attribute__((ext_vector_type(4)))  unsigned uv4;
typedef __attribute__((ext_vector_type(4)))  int      iv4;
typedef __attribute__((ext_vector_type(8)))  int      iv8;

static constexpr int Bn = 4, Tn = 512, Fn = 352, Fp = 88;

__constant__ int c_dil[8] = {48, 76, 96, 111, 124, 135, 144, 152};

// ---------------------------------------------------------------------------
// Tensor Data Mover helpers (gfx1250). Guarded so the host pass and builds
// without the builtin still compile (falls back to global-gather staging).
// ---------------------------------------------------------------------------
#if defined(__AMDGCN__) && __has_builtin(__builtin_amdgcn_tensor_load_to_lds)
#define HAVE_TDM 1
#else
#define HAVE_TDM 0
#endif

// 2D tile load: tile_dim0 = TILE0 elements along dim0 (contiguous f32),
// tile_dim1 = 32 rows with stride `stride0` elements. D# per ISA §8.3/8.4.
template <int TILE0>
__device__ inline void tdm_issue2d(const float* gptr, const void* ldsptr,
                                   int tdim0, long long stride0) {
#if HAVE_TDM
  unsigned long long ga = (unsigned long long)(size_t)gptr;
  unsigned lds = (unsigned)(size_t)ldsptr;      // LDS aperture: addr[31:0]
  uv4 g0;
  g0[0] = 1u;                                   // count=1, user descriptor
  g0[1] = lds;                                  // lds_addr
  g0[2] = (unsigned)(ga & 0xFFFFFFFFull);       // global_addr[31:0]
  g0[3] = ((unsigned)(ga >> 32) & 0x01FFFFFFu)  // global_addr[56:32]
          | (2u << 30);                         // type=2 ("image")
  iv8 g1;
  g1[0] = (int)(2u << 16);                      // data_size = 4 bytes
  g1[1] = (int)(((unsigned)tdim0 & 0xFFFFu) << 16);          // tensor_dim0 lo
  g1[2] = (int)((((unsigned)tdim0 >> 16) & 0xFFFFu)          // tensor_dim0 hi
                | (128u << 16));                             // tensor_dim1=128
  g1[3] = (int)((unsigned)TILE0 << 16);                      // tile_dim0
  g1[4] = 32;                                                // tile_dim1 = 32
  g1[5] = (int)((unsigned long long)stride0 & 0xFFFFFFFFull);// dim0 stride lo
  g1[6] = (int)(((unsigned long long)stride0 >> 32) & 0xFFFFull);
  g1[7] = 0;
  iv4 z4 = {0, 0, 0, 0};
#if __clang_major__ >= 23
  iv8 z8 = {0, 0, 0, 0, 0, 0, 0, 0};
  __builtin_amdgcn_tensor_load_to_lds(g0, g1, z4, z4, z8, 0);
#else
  __builtin_amdgcn_tensor_load_to_lds(g0, g1, z4, z4, 0);
#endif
#else
  (void)gptr; (void)ldsptr; (void)tdim0; (void)stride0;
#endif
}

__device__ inline void tensor_wait0() {
#if HAVE_TDM
  __builtin_amdgcn_s_wait_tensorcnt(0);
#endif
}
__device__ inline void tensor_wait1() {
#if HAVE_TDM
  __builtin_amdgcn_s_wait_tensorcnt(1);
#endif
}

// ---------------------------------------------------------------------------
// Weight repack: float source -> f16 in WMMA *fragment order*:
//   dst[(chunk*WT + oc/16)*512 + lane*16 + i]
// so each lane of a wave loads its v16h B fragment as one contiguous,
// coalesced 32B global read (weights stay resident in WGP$/L2).
// mode_hd=0: src [Cout][Cin][ntaps]; mode_hd=1: src [8][Cout][Cin][3].
// ---------------------------------------------------------------------------
__global__ void k_pack_w(const float* __restrict__ wsrc, _Float16* __restrict__ wdst,
                         int Cin, int Cout, int ntaps, int Kpad, int mode_hd) {
  int idx = blockIdx.x * blockDim.x + threadIdx.x;
  int total = Kpad * Cout;
  if (idx >= total) return;
  int k = idx / Cout, oc = idx - k * Cout;
  float v = 0.f;
  int K = ntaps * Cin;
  if (k < K) {
    int tap = k / Cin, ic = k - tap * Cin;
    size_t src;
    if (mode_hd) {
      int di = tap / 3, j = tap - di * 3;
      src = (((size_t)di * Cout + oc) * Cin + ic) * 3 + j;
    } else {
      src = ((size_t)oc * Cin + ic) * ntaps + tap;
    }
    v = wsrc[src];
  }
  int WT    = Cout >> 4;
  int chunk = k >> 5, kin = k & 31;
  int halfb = (kin >> 3) & 1;
  int i     = (kin & 7) + ((kin >> 4) << 3);
  int lane  = (oc & 15) + (halfb << 4);
  wdst[(((size_t)chunk * WT + (oc >> 4)) << 9) + lane * 16 + i] = (_Float16)v;
}

// sum the 8 per-branch bias vectors of the harmonic-dilated stack
__global__ void k_sum_bias(const float* __restrict__ b, float* __restrict__ o) {
  int oc = threadIdx.x;
  float s = 0.f;
  for (int i = 0; i < 8; ++i) s += b[i * 128 + oc];
  o[oc] = s;
}

// ---------------------------------------------------------------------------
// Implicit-GEMM conv via V_WMMA_F32_16X16X32_F16 (templated layer config).
//   M = F positions (16/tile), N = out channels (16/tile), K = taps*CIN.
//   8 waves/block: wave -> (N-tile wv%WT, ACC consecutive M-tiles).
//   B fragments: direct coalesced global loads of the fragment-ordered pack.
//   A tiles: CIN==128 + TDM available -> per-chunk TENSOR_LOAD_TO_LDS of the
//   raw 32ch x MB tile (double-buffered, TENSORcnt-pipelined 2 deep), then an
//   LDS->LDS fragment repack; otherwise cooperative global gather.
// Epilogue: +bias, optional ReLU, optional fused MaxPool(1,4).
// ---------------------------------------------------------------------------
template <int CIN, int KT, int KF, int DILT, int DILF, int MODE_HD,
          int WT, int ACC, int RELU, int POOL, int USE_TDM>
__global__ __launch_bounds__(256)
void k_conv_wmma(const float* __restrict__ x, const _Float16* __restrict__ wp,
                 const float* __restrict__ bias, float* __restrict__ out,
                 int T, int F, int Fout) {
  constexpr int NTAPS = MODE_HD ? 24 : KT * KF;
  constexpr int K     = NTAPS * CIN;
  constexpr int KPAD  = (K + 31) & ~31;
  constexpr int NCH   = KPAD >> 5;
  constexpr int COUT  = WT * 16;
  constexpr int MTT   = (8 / WT) * ACC;  // M-tiles per block
  constexpr int MB    = MTT * 16;        // F extent per block
  constexpr int NA    = MTT * 512;       // staged A elements per chunk
  constexpr bool TDM  = (USE_TDM != 0) && (HAVE_TDM != 0) && (CIN == 128);

  __shared__ __align__(32) _Float16 As[MTT][512];
  __shared__ __align__(16) float raw[TDM ? 2 * 32 * MB : 1];  // TDM landing pad

  const int tid  = threadIdx.x;
  const int lane = tid & 31;
  const int wv   = tid >> 5;
  const int half = lane >> 4;
  const int l16  = lane & 15;

  const int fblk = blockIdx.x * MB;
  const int t    = blockIdx.y;
  const int n    = blockIdx.z;

  const int my_wt = wv % WT;
  const int mt0   = (wv / WT) * ACC;

  v8f acc[ACC] = {};

  const _Float16* wpf = wp + ((size_t)my_wt << 9) + lane * 16;

  // per-tap geometry (also valid per chunk when CIN==128: tap = kc>>2)
  auto tap_geom = [&](int tap, int& dt, int& df) {
    if (MODE_HD) {
      int di = tap / 3, jj = tap - di * 3;
      dt = 0; df = (jj - 1) * c_dil[di];
    } else {
      int kh = tap / KF, kw = tap - kh * KF;
      dt = (kh - KT / 2) * DILT;
      df = (kw - KF / 2) * DILF;
    }
  };

  if constexpr (TDM) {
    auto issue = [&](int c) {
      int dt, df;
      tap_geom(c >> 2, dt, df);
      int tc = t + dt;
      tc = tc < 0 ? 0 : (tc > T - 1 ? T - 1 : tc);
      const float* gp =
          x + (((size_t)n * CIN + (size_t)(c & 3) * 32) * T + tc) * F + (fblk + df);
      tdm_issue2d<MB>(gp, &raw[(c & 1) * 32 * MB], F, (long long)T * F);
    };
    if (wv == 0) {
      issue(0);
      if (NCH > 1) issue(1);
    }
    for (int kc = 0; kc < NCH; ++kc) {
      if (wv == 0) {
        if (kc + 1 < NCH) tensor_wait1();   // oldest (kc) complete
        else              tensor_wait0();
      }
      __syncthreads();
      // LDS raw tile -> fragment-order As (f32 -> f16), zero padding applied
      int dt, df;
      tap_geom(kc >> 2, dt, df);
      const int tc = t + dt;
      const float* rb = &raw[(kc & 1) * 32 * MB];
#pragma unroll
      for (int j = 0; j < NA / 256; ++j) {
        const int s    = tid + j * 256;
        const int tile = s >> 9;
        const int li   = s & 511;
        const int ln   = li >> 4;
        const int i    = li & 15;
        const int hlf  = ln >> 4;
        const int kin  = ((i >> 3) << 4) + (i & 7) + (hlf << 3); // = local ic
        const int m    = (tile << 4) + (ln & 15);
        const int fc   = fblk + m + df;
        _Float16 v = (_Float16)0.f;
        if (tc >= 0 && tc < T && fc >= 0 && fc < F)
          v = (_Float16)rb[kin * MB + m];
        As[tile][li] = v;
      }
      __syncthreads();
      if (wv == 0 && kc + 2 < NCH) issue(kc + 2);   // refill just-freed buffer
      v16h b = *(const v16h*)&wpf[(size_t)kc * WT * 512];
#pragma unroll
      for (int a = 0; a < ACC; ++a) {
        v16h av = *(const v16h*)&As[mt0 + a][lane << 4];
        acc[a] = __builtin_amdgcn_wmma_f32_16x16x32_f16(
            false, av, false, b, (short)0, acc[a], false, false);
      }
    }
  } else {
    for (int kc = 0; kc < NCH; ++kc) {
      // cooperative A gather (f32 -> f16, fragment order)
#pragma unroll
      for (int j = 0; j < NA / 256; ++j) {
        const int s    = tid + j * 256;
        const int tile = s >> 9;
        const int li   = s & 511;
        const int ln   = li >> 4;
        const int i    = li & 15;
        const int hlf  = ln >> 4;
        const int kin  = ((i >> 3) << 4) + (i & 7) + (hlf << 3);
        const int k    = (kc << 5) + kin;
        _Float16 v = (_Float16)0.f;
        if (k < K) {
          const int tap = k / CIN;           // shift (CIN pow2 or 1)
          const int ic  = k - tap * CIN;
          int dt, df;
          tap_geom(tap, dt, df);
          const int m  = (tile << 4) + (ln & 15);
          const int tc = t + dt;
          const int fc = fblk + m + df;
          if (tc >= 0 && tc < T && fc >= 0 && fc < F)
            v = (_Float16)x[(((size_t)n * CIN + ic) * T + tc) * F + fc];
        }
        As[tile][li] = v;
      }
      __syncthreads();
      v16h b = *(const v16h*)&wpf[(size_t)kc * WT * 512];
      if (kc + 1 < NCH)
        __builtin_prefetch(&wpf[(size_t)(kc + 1) * WT * 512], 0, 2);
#pragma unroll
      for (int a = 0; a < ACC; ++a) {
        v16h av = *(const v16h*)&As[mt0 + a][lane << 4];
        acc[a] = __builtin_amdgcn_wmma_f32_16x16x32_f16(
            false, av, false, b, (short)0, acc[a], false, false);
      }
      __syncthreads();
    }
  }

  // ---- epilogue: C fragment lane holds N=oc, VGPR r -> M = r + half*8 ----
  const int oc = (my_wt << 4) + l16;
  const float bz = bias[oc];
  const size_t obase = (((size_t)n * COUT + oc) * T + t) * Fout;
#pragma unroll
  for (int a = 0; a < ACC; ++a) {
    float v[8];
#pragma unroll
    for (int r = 0; r < 8; ++r) {
      float u = acc[a][r] + bz;
      if (RELU) u = fmaxf(u, 0.f);
      v[r] = u;
    }
    if (!POOL) {
#pragma unroll
      for (int r = 0; r < 8; ++r) {
        int f = fblk + ((mt0 + a) << 4) + r + (half << 3);
        if (f < Fout) out[obase + f] = v[r];
      }
    } else {
      float p0 = fmaxf(fmaxf(v[0], v[1]), fmaxf(v[2], v[3]));
      float p1 = fmaxf(fmaxf(v[4], v[5]), fmaxf(v[6], v[7]));
      int f0 = ((fblk + ((mt0 + a) << 4)) >> 2) + (half << 1);
      if (f0 < Fout)     out[obase + f0]     = p0;
      if (f0 + 1 < Fout) out[obase + f0 + 1] = p1;
    }
  }
}

// ---------------------------------------------------------------------------
// InstanceNorm2d (affine=False): one block per (n,c) plane, in place.
// ---------------------------------------------------------------------------
__global__ __launch_bounds__(256)
void k_inorm(float* __restrict__ x, int HW) {
  __shared__ float rs[256], rq[256];
  const size_t base = (size_t)blockIdx.x * HW;
  float s = 0.f, q = 0.f;
  for (int i = threadIdx.x; i < HW; i += 256) {
    float v = x[base + i];
    s += v; q += v * v;
  }
  rs[threadIdx.x] = s; rq[threadIdx.x] = q;
  __syncthreads();
  for (int off = 128; off > 0; off >>= 1) {
    if (threadIdx.x < off) {
      rs[threadIdx.x] += rs[threadIdx.x + off];
      rq[threadIdx.x] += rq[threadIdx.x + off];
    }
    __syncthreads();
  }
  const float inv  = 1.f / (float)HW;
  const float mean = rs[0] * inv;
  const float var  = rq[0] * inv - mean * mean;
  const float rstd = rsqrtf(var + 1e-5f);
  for (int i = threadIdx.x; i < HW; i += 256)
    x[base + i] = (x[base + i] - mean) * rstd;
}

// ---------------------------------------------------------------------------
template <int CIN, int KT, int KF, int DILT, int DILF, int MODE_HD,
          int WT, int ACC, int RELU, int POOL, int USE_TDM>
static void run_conv(const float* xin, const float* w, const float* bias,
                     float* outp, _Float16* wpack, int F_in, int F_out,
                     hipStream_t stream) {
  constexpr int NTAPS = MODE_HD ? 24 : KT * KF;
  constexpr int K     = NTAPS * CIN;
  constexpr int KPAD  = (K + 31) & ~31;
  constexpr int COUT  = WT * 16;
  constexpr int MB    = (8 / WT) * ACC * 16;
  int npack = KPAD * COUT;
  k_pack_w<<<dim3((npack + 255) / 256), dim3(256), 0, stream>>>(
      w, wpack, CIN, COUT, NTAPS, KPAD, MODE_HD);
  dim3 g((F_in + MB - 1) / MB, Tn, Bn);
  k_conv_wmma<CIN, KT, KF, DILT, DILF, MODE_HD, WT, ACC, RELU, POOL, USE_TDM>
      <<<g, dim3(256), 0, stream>>>(xin, wpack, bias, outp, Tn, F_in, F_out);
}

extern "C" void kernel_launch(void* const* d_in, const int* in_sizes, int n_in,
                              void* d_out, int out_size, void* d_ws, size_t ws_size,
                              hipStream_t stream) {
  (void)in_sizes; (void)n_in; (void)out_size; (void)ws_size;

  const float* cqt  = (const float*)d_in[0];
  const float* w0   = (const float*)d_in[1];  const float* b0  = (const float*)d_in[2];
  const float* w1   = (const float*)d_in[3];  const float* b1  = (const float*)d_in[4];
  const float* w2   = (const float*)d_in[5];  const float* b2  = (const float*)d_in[6];
  const float* w_hd = (const float*)d_in[7];  const float* bhd = (const float*)d_in[8];
  const float* wd0  = (const float*)d_in[9];  const float* bd0 = (const float*)d_in[10];
  const float* wd1  = (const float*)d_in[11]; const float* bd1 = (const float*)d_in[12];
  const float* wd2  = (const float*)d_in[13]; const float* bd2 = (const float*)d_in[14];
  const float* wd3  = (const float*)d_in[15]; const float* bd3 = (const float*)d_in[16];
  const float* wd4  = (const float*)d_in[17]; const float* bd4 = (const float*)d_in[18];

  size_t o = 0;
  auto take = [&](size_t bytes) -> void* {
    o = (o + 255) & ~(size_t)255;
    void* p = (char*)d_ws + o;
    o += bytes;
    return p;
  };
  float*    bufA   = (float*)take((size_t)Bn * 16 * Tn * Fn * 4);
  float*    bufB   = (float*)take((size_t)Bn * 16 * Tn * Fn * 4);
  float*    bufY   = (float*)take((size_t)Bn * 128 * Tn * Fn * 4);
  float*    bufZ1  = (float*)take((size_t)Bn * 128 * Tn * Fp * 4);
  float*    bufZ2  = (float*)take((size_t)Bn * 128 * Tn * Fp * 4);
  _Float16* wpack  = (_Float16*)take(1 << 20);
  float*    hdbias = (float*)take(512);

  auto inorm = [&](float* xp, int C, int HW) {
    k_inorm<<<dim3(Bn * C), dim3(256), 0, stream>>>(xp, HW);
  };

  // cnn: 3 x (conv7x7 -> relu -> instancenorm)
  run_conv<1, 7, 7, 1, 1, 0, 1, 1, 1, 0, 0>(cqt, w0, b0, bufA, wpack, Fn, Fn, stream);
  inorm(bufA, 16, Tn * Fn);
  run_conv<16, 7, 7, 1, 1, 0, 1, 1, 1, 0, 0>(bufA, w1, b1, bufB, wpack, Fn, Fn, stream);
  inorm(bufB, 16, Tn * Fn);
  run_conv<16, 7, 7, 1, 1, 0, 1, 1, 1, 0, 0>(bufB, w2, b2, bufA, wpack, Fn, Fn, stream);
  inorm(bufA, 16, Tn * Fn);

  // harmonic dilated stack: 24 taps (8 dilations x 3), summed biases
  k_sum_bias<<<dim3(1), dim3(128), 0, stream>>>(bhd, hdbias);
  run_conv<16, 1, 3, 1, 1, 1, 8, 4, 0, 0, 0>(bufA, w_hd, hdbias, bufY, wpack, Fn, Fn, stream);

  // d_cnn (128-channel layers: TDM-staged activations)
  run_conv<128, 1, 3, 1, 48, 0, 8, 4, 1, 1, 1>(bufY, wd0, bd0, bufZ1, wpack, Fn, Fp, stream);
  inorm(bufZ1, 128, Tn * Fp);
  run_conv<128, 1, 3, 1, 12, 0, 8, 2, 1, 0, 1>(bufZ1, wd1, bd1, bufZ2, wpack, Fp, Fp, stream);
  inorm(bufZ2, 128, Tn * Fp);
  run_conv<128, 5, 1, 1, 1, 0, 8, 2, 1, 0, 1>(bufZ2, wd2, bd2, bufZ1, wpack, Fp, Fp, stream);
  inorm(bufZ1, 128, Tn * Fp);
  run_conv<128, 5, 1, 1, 1, 0, 8, 2, 1, 0, 1>(bufZ1, wd3, bd3, bufZ2, wpack, Fp, Fp, stream);
  inorm(bufZ2, 128, Tn * Fp);
  float* zout = (float*)d_out;
  run_conv<128, 5, 1, 1, 1, 0, 8, 2, 1, 0, 1>(bufZ2, wd4, bd4, zout, wpack, Fp, Fp, stream);
  inorm(zout, 128, Tn * Fp);
}
